// GAT_35321811042915
// MI455X (gfx1250) — matrix operand order repaired and checked by reading, not compiled, not run on hardware
//
#include <hip/hip_runtime.h>

typedef __attribute__((ext_vector_type(16))) _Float16 v16h;
typedef __attribute__((ext_vector_type(8)))  _Float16 v8h;
typedef __attribute__((ext_vector_type(4)))  _Float16 v4h;
typedef __attribute__((ext_vector_type(8)))  float    v8f;

#define NEG_SLOPE 0.2f

// ---- order-preserving float<->uint encoding for integer atomicMax ----
__device__ __forceinline__ unsigned fenc(float f) {
  unsigned u = __float_as_uint(f);
  return (u & 0x80000000u) ? ~u : (u | 0x80000000u);
}
__device__ __forceinline__ float fdec(unsigned u) {
  unsigned v = (u & 0x80000000u) ? (u & 0x7fffffffu) : ~u;
  return __uint_as_float(v);
}
#define ENC_NEG_INF 0x007fffffu   // encoded(-inf): minimum encoding

// ---- native f32 atomic add (CDNA5 global_atomic_add_f32, no CAS expansion) ----
__device__ __forceinline__ void atomAddF32(float* p, float v) {
  asm volatile("global_atomic_add_f32 %0, %1, off" : : "v"(p), "v"(v) : "memory");
}

// ---------------- elementwise helpers (vectorized) ----------------
__global__ void k_f2h4(const float* __restrict__ in, _Float16* __restrict__ out, unsigned n4) {
  unsigned t = blockIdx.x * blockDim.x + threadIdx.x;
  if (t >= n4) return;
  float4 v = ((const float4*)in)[t];
  v4h o = { (_Float16)v.x, (_Float16)v.y, (_Float16)v.z, (_Float16)v.w };
  ((v4h*)out)[t] = o;
}

__global__ void k_fill4(uint4* __restrict__ p, unsigned v, unsigned n4) {
  unsigned t = blockIdx.x * blockDim.x + threadIdx.x;
  if (t < n4) p[t] = make_uint4(v, v, v, v);
}

__global__ void k_fill_u32(unsigned* __restrict__ p, unsigned v, unsigned n) {
  unsigned t = blockIdx.x * blockDim.x + threadIdx.x;
  if (t < n) p[t] = v;
}

// ---------------- WMMA GEMM: C[MxNout] = A[MxK] * B[NoutxK]^T  (f16 in, f32 out) ----
// One wave computes a 16x32 C tile (two 16x16 WMMA tiles sharing the A fragment).
__global__ __launch_bounds__(256)
void k_gemm_f16(const _Float16* __restrict__ A, const _Float16* __restrict__ B,
                float* __restrict__ C, int rowTiles, int colPairs,
                long long M, int K, int Nout) {
  int lane = threadIdx.x & 31;
  int tile = blockIdx.x * (blockDim.x >> 5) + (threadIdx.x >> 5);
  if (tile >= rowTiles * colPairs) return;
  int rt = tile / colPairs, cp = tile % colPairs;
  int ln = lane & 15;
  int hi = lane >> 4;                 // 0 = lanes 0-15, 1 = lanes 16-31

  long long arowi = (long long)rt * 16 + ln;
  if (arowi >= M) arowi = M - 1;      // clamp for ragged M (EXEC stays all-1 for WMMA)
  const _Float16* arow  = A + (size_t)arowi * K;
  const _Float16* brow0 = B + (size_t)(cp * 32 + ln) * K;
  const _Float16* brow1 = brow0 + (size_t)16 * K;

  const int ka = hi ? 16 : 0;         // A frag: 16 contiguous halves at +ka
  const int kb = hi ? 16 : 0;         // B frag: 16 contiguous halves at +kb

  v8f acc0 = {}, acc1 = {};
  for (int k0 = 0; k0 < K; k0 += 32) {
    v8h alo  = *(const v8h*)(arow  + k0 + ka);
    v8h ahi  = *(const v8h*)(arow  + k0 + ka + 8);
    v8h b0lo = *(const v8h*)(brow0 + k0 + kb);
    v8h b0hi = *(const v8h*)(brow0 + k0 + kb + 8);
    v8h b1lo = *(const v8h*)(brow1 + k0 + kb);
    v8h b1hi = *(const v8h*)(brow1 + k0 + kb + 8);
    v16h av, b0, b1;
#pragma unroll
    for (int i = 0; i < 8; ++i) {
      av[i] = alo[i];  av[i + 8] = ahi[i];
      b0[i] = b0lo[i]; b0[i + 8] = b0hi[i];
      b1[i] = b1lo[i]; b1[i + 8] = b1hi[i];
    }
    acc0 = __builtin_amdgcn_wmma_f32_16x16x32_f16(false, av, false, b0, (short)0, acc0, false, false);
    acc1 = __builtin_amdgcn_wmma_f32_16x16x32_f16(false, av, false, b1, (short)0, acc1, false, false);
  }

  int mb = hi ? 8 : 0;                // C: vgpr r -> row r+mb, col = ln
  long long r0 = (long long)rt * 16 + mb;
#pragma unroll
  for (int r = 0; r < 8; ++r) {
    long long row = r0 + r;
    if (row < M) {
      float* c = C + row * Nout + cp * 32 + ln;
      c[0]  = acc0[r];
      c[16] = acc1[r];
    }
  }
}

// ---------------- per-node attention coefficients (C fixed at 64) ----------------
// as[n*H+h] = dot(Hp[n, h*64 : h*64+64], att_s[h]);  ad likewise
__global__ void k_att_node(const float* __restrict__ Hp,
                           const float* __restrict__ ats, const float* __restrict__ atd,
                           float* __restrict__ as, float* __restrict__ ad,
                           unsigned N, int hShift) {
  unsigned t = blockIdx.x * blockDim.x + threadIdx.x;
  if (t >= (N << hShift)) return;
  unsigned n = t >> hShift;
  unsigned h = t & ((1u << hShift) - 1);
  const float4* row = (const float4*)(Hp + (((size_t)n << hShift) << 6) + (h << 6));
  const float4* vs  = (const float4*)(ats + (h << 6));
  const float4* vd  = (const float4*)(atd + (h << 6));
  float s = 0.f, d = 0.f;
#pragma unroll
  for (int c = 0; c < 16; ++c) {
    float4 r = row[c], a = vs[c], b = vd[c];
    s += r.x * a.x + r.y * a.y + r.z * a.z + r.w * a.w;
    d += r.x * b.x + r.y * b.y + r.z * b.z + r.w * b.w;
  }
  as[t] = s; ad[t] = d;
}

// ---------------- edge kernels (edges 0..E-1 then N self-loops) ----------------
__device__ __forceinline__ void edge_sd(const long long* __restrict__ ei,
                                        unsigned e, unsigned E, unsigned& s, unsigned& d) {
  if (e < E) { s = (unsigned)ei[e]; d = (unsigned)ei[E + e]; } else { s = d = e - E; }
}

__global__ void k_edge_max(const long long* __restrict__ ei,
                           const float* __restrict__ as, const float* __restrict__ ad,
                           unsigned* __restrict__ m,
                           unsigned E, unsigned NE, int hShift) {
  unsigned t = blockIdx.x * blockDim.x + threadIdx.x;
  if (t >= (NE << hShift)) return;
  unsigned e = t >> hShift, h = t & ((1u << hShift) - 1);
  unsigned s, d; edge_sd(ei, e, E, s, d);
  float z = as[(s << hShift) + h] + ad[(d << hShift) + h];
  z = z > 0.f ? z : NEG_SLOPE * z;
  atomicMax(&m[(d << hShift) + h], fenc(z));
}

__global__ void k_edge_exp(const long long* __restrict__ ei,
                           const float* __restrict__ as, const float* __restrict__ ad,
                           const unsigned* __restrict__ m,
                           float* __restrict__ ex, float* __restrict__ den,
                           unsigned E, unsigned NE, int hShift) {
  unsigned t = blockIdx.x * blockDim.x + threadIdx.x;
  if (t >= (NE << hShift)) return;
  unsigned e = t >> hShift, h = t & ((1u << hShift) - 1);
  unsigned s, d; edge_sd(ei, e, E, s, d);
  float z = as[(s << hShift) + h] + ad[(d << hShift) + h];
  z = z > 0.f ? z : NEG_SLOPE * z;
  float v = expf(z - fdec(m[(d << hShift) + h]));
  ex[t] = v;
  atomAddF32(&den[(d << hShift) + h], v);
}

// out[dst, f] += alpha(e,h) * Hp[src, f]   with f = h*64 + c, F = H*64 (pow2)
__global__ void k_edge_aggr(const long long* __restrict__ ei,
                            const float* __restrict__ ex, const float* __restrict__ den,
                            const float* __restrict__ Hp, float* __restrict__ out,
                            unsigned E, unsigned NE, int hShift) {
  const int fShift = hShift + 6;
  unsigned t = blockIdx.x * blockDim.x + threadIdx.x;   // max NE<<8 ~ 218M < 2^31
  if (t >= (NE << fShift)) return;
  unsigned e = t >> fShift, f = t & ((1u << fShift) - 1);
  unsigned h = f >> 6;
  unsigned s, d; edge_sd(ei, e, E, s, d);
  float alpha = ex[(e << hShift) + h] / (den[(d << hShift) + h] + 1e-16f);
  atomAddF32(out + (((size_t)d) << fShift) + f, alpha * Hp[(((size_t)s) << fShift) + f]);
}

// h = relu(h + b); optional f16 copy for the next WMMA GEMM (vectorized by 4, F pow2)
__global__ void k_bias_relu4(float* __restrict__ h, const float* __restrict__ b,
                             _Float16* __restrict__ hh, unsigned total4, int F) {
  unsigned t = blockIdx.x * blockDim.x + threadIdx.x;
  if (t >= total4) return;
  float4 v = ((const float4*)h)[t];
  float4 bb = ((const float4*)b)[t & ((unsigned)(F >> 2) - 1)];
  v.x = fmaxf(v.x + bb.x, 0.f); v.y = fmaxf(v.y + bb.y, 0.f);
  v.z = fmaxf(v.z + bb.z, 0.f); v.w = fmaxf(v.w + bb.w, 0.f);
  ((float4*)h)[t] = v;
  if (hh) {
    v4h o = { (_Float16)v.x, (_Float16)v.y, (_Float16)v.z, (_Float16)v.w };
    ((v4h*)hh)[t] = o;
  }
}

// ---------------- root selection + final FC ----------------
__global__ void k_find_root(const float* __restrict__ x, int* __restrict__ root,
                            unsigned N, int stride) {
  unsigned n = blockIdx.x * blockDim.x + threadIdx.x;
  if (n < N && x[(size_t)n * stride] == 0.f) atomicMin(root, (int)n);
}

__global__ void k_fc_root(const float* __restrict__ h2, const float* __restrict__ W,
                          const float* __restrict__ bias, const int* __restrict__ root,
                          float* __restrict__ out) {
  int c = threadIdx.x;                   // 64 threads
  long long r = *root;
  const float* hr = h2 + r * 64;
  float acc = bias[c];
#pragma unroll
  for (int k = 0; k < 64; ++k) acc += hr[k] * W[c * 64 + k];
  out[c] = acc;
}

// ---------------- host orchestration ----------------
extern "C" void kernel_launch(void* const* d_in, const int* in_sizes, int n_in,
                              void* d_out, int out_size, void* d_ws, size_t ws_size,
                              hipStream_t stream) {
  const float*     x     = (const float*)d_in[0];
  const long long* ei    = (const long long*)d_in[1];   // int64 [2,E]
  const float*     W1    = (const float*)d_in[2];
  const float*     att1s = (const float*)d_in[3];
  const float*     att1d = (const float*)d_in[4];
  const float*     b1    = (const float*)d_in[5];
  const float*     W2    = (const float*)d_in[6];
  const float*     att2s = (const float*)d_in[7];
  const float*     att2d = (const float*)d_in[8];
  const float*     b2    = (const float*)d_in[9];
  const float*     Wfc   = (const float*)d_in[10];
  const float*     bfc   = (const float*)d_in[11];
  float* out = (float*)d_out;

  const int IN = 128, H1 = 4, F1 = 256, F2 = 64;
  const int hS1 = 2, hS2 = 0;                 // log2(heads)
  const unsigned N  = (unsigned)(in_sizes[0] / IN);
  const unsigned E  = (unsigned)(in_sizes[1] / 2);
  const unsigned NE = E + N;

  // carve workspace (256B aligned)
  char* p = (char*)d_ws;
  auto carve = [&](size_t bytes) -> void* {
    void* r = (void*)p; p += (bytes + 255) & ~(size_t)255; return r;
  };
  _Float16* xh  = (_Float16*)carve((size_t)N * IN * 2);          // reused as h2 (f32, same bytes)
  _Float16* w1h = (_Float16*)carve((size_t)F1 * IN * 2);
  _Float16* w2h = (_Float16*)carve((size_t)F2 * F1 * 2);
  float*    hp1 = (float*)carve((size_t)N * F1 * 4);             // reused as h1h (f16, half)
  float*    as1 = (float*)carve((size_t)N * H1 * 4);
  float*    ad1 = (float*)carve((size_t)N * H1 * 4);
  unsigned* m1  = (unsigned*)carve((size_t)N * H1 * 4);
  float*    den1= (float*)carve((size_t)N * H1 * 4);
  size_t ex1b = (size_t)NE * H1 * 4, hp2b = (size_t)N * F2 * 4;
  float*    ex1 = (float*)carve(ex1b > hp2b ? ex1b : hp2b);      // reused as hp2
  float*    h1  = (float*)carve((size_t)N * F1 * 4);
  float*    as2 = (float*)carve((size_t)N * 4);
  float*    ad2 = (float*)carve((size_t)N * 4);
  unsigned* m2  = (unsigned*)carve((size_t)N * 4);
  float*    den2= (float*)carve((size_t)N * 4);
  float*    ex2 = (float*)carve((size_t)NE * 4);
  int*      root= (int*)carve(256);

  _Float16* h1h = (_Float16*)hp1;   // valid after hp1 is dead
  float*    hp2 = (float*)ex1;      // valid after ex1 is dead
  float*    h2  = (float*)xh;       // valid after xh is dead

  const int BS = 256;
  auto nb = [&](long long total) { return (unsigned)((total + BS - 1) / BS); };

  // 1. casts to f16 (vectorized x4)
  k_f2h4<<<nb((long long)N * IN / 4), BS, 0, stream>>>(x, xh, N * IN / 4);
  k_f2h4<<<nb(F1 * IN / 4), BS, 0, stream>>>(W1, w1h, F1 * IN / 4);
  k_f2h4<<<nb(F2 * F1 / 4), BS, 0, stream>>>(W2, w2h, F2 * F1 / 4);

  // 2. GEMM1: hp1[N,256] = xh @ W1^T    (WMMA f16->f32, 16x32 tile/wave)
  {
    int rowT = (int)((N + 15) / 16), cPair = F1 / 32;
    k_gemm_f16<<<nb((long long)rowT * cPair * 32), BS, 0, stream>>>(
        xh, w1h, hp1, rowT, cPair, N, IN, F1);
  }

  // 3. layer-1 attention + softmax-aggregate over edges
  k_att_node<<<nb((long long)N * H1), BS, 0, stream>>>(hp1, att1s, att1d, as1, ad1, N, hS1);
  k_fill4<<<nb((long long)N * H1 / 4), BS, 0, stream>>>((uint4*)m1, ENC_NEG_INF, N * H1 / 4);
  k_fill4<<<nb((long long)N * H1 / 4), BS, 0, stream>>>((uint4*)den1, 0u, N * H1 / 4);
  k_fill4<<<nb((long long)N * F1 / 4), BS, 0, stream>>>((uint4*)h1, 0u, N * F1 / 4);
  k_edge_max<<<nb((long long)NE << hS1), BS, 0, stream>>>(ei, as1, ad1, m1, E, NE, hS1);
  k_edge_exp<<<nb((long long)NE << hS1), BS, 0, stream>>>(ei, as1, ad1, m1, ex1, den1, E, NE, hS1);
  k_edge_aggr<<<nb((long long)NE << (hS1 + 6)), BS, 0, stream>>>(ei, ex1, den1, hp1, h1, E, NE, hS1);
  k_bias_relu4<<<nb((long long)N * F1 / 4), BS, 0, stream>>>(h1, b1, h1h, N * F1 / 4, F1);

  // 4. GEMM2: hp2[N,64] = h1h @ W2^T
  {
    int rowT = (int)((N + 15) / 16), cPair = F2 / 32;
    k_gemm_f16<<<nb((long long)rowT * cPair * 32), BS, 0, stream>>>(
        h1h, w2h, hp2, rowT, cPair, N, F1, F2);
  }

  // 5. layer-2 attention (H=1)
  k_att_node<<<nb(N), BS, 0, stream>>>(hp2, att2s, att2d, as2, ad2, N, hS2);
  k_fill4<<<nb(N / 4 + 1), BS, 0, stream>>>((uint4*)m2, ENC_NEG_INF, N / 4);
  k_fill4<<<nb(N / 4 + 1), BS, 0, stream>>>((uint4*)den2, 0u, N / 4);
  k_fill4<<<nb((long long)N * F2 / 4), BS, 0, stream>>>((uint4*)h2, 0u, N * F2 / 4);
  k_edge_max<<<nb(NE), BS, 0, stream>>>(ei, as2, ad2, m2, E, NE, hS2);
  k_edge_exp<<<nb(NE), BS, 0, stream>>>(ei, as2, ad2, m2, ex2, den2, E, NE, hS2);
  k_edge_aggr<<<nb((long long)NE << 6), BS, 0, stream>>>(ei, ex2, den2, hp2, h2, E, NE, hS2);
  k_bias_relu4<<<nb((long long)N * F2 / 4), BS, 0, stream>>>(h2, b2, (_Float16*)nullptr, N * F2 / 4, F2);

  // 6. root readout
  k_fill_u32<<<1, 1, 0, stream>>>((unsigned*)root, N, 1);
  k_find_root<<<nb(N), BS, 0, stream>>>(x, root, N, IN);
  k_fc_root<<<1, 64, 0, stream>>>(h2, Wfc, bfc, root, out);
}
